// TCMlpDWBN_81990925681417
// MI455X (gfx1250) — matrix operand order repaired and checked by reading, not compiled, not run on hardware
//
#include <hip/hip_runtime.h>
#include <hip/hip_bf16.h>

typedef __attribute__((ext_vector_type(16))) _Float16 v16h;
typedef __attribute__((ext_vector_type(8)))  float    v8f;

#define BATCH   4
#define NTOK    3136
#define NPTS    12544          // N0 points per batch image
#define RROWS   12544          // BATCH*NTOK token rows
#define HWPIX   12544          // 112*112
#define W_      112
#define CIN     128
#define CHID    512
#define COUT    128
#define EPSF    1e-6f
#define BNEPS   1e-5f
#define INV_R   (1.0f/12544.0f)

// ---------------- packing into WMMA register order ----------------
// A (16-bit 16x32, ISA 7.12.2): lane l: M=l%16, hi=l/16; elem e: K = 16*(e/8)+8*hi+(e%8)
__global__ void pack_a_kernel(const float* __restrict__ Wm, _Float16* __restrict__ Ap,
                              int KC, int K, int total) {
    int t = blockIdx.x * blockDim.x + threadIdx.x;
    if (t >= total) return;
    int e = t & 15, l = (t >> 4) & 31, rest = t >> 9;
    int kc = rest % KC, mt = rest / KC;
    int m  = mt * 16 + (l & 15);
    int hi = l >> 4;
    int k  = kc * 32 + 16 * (e >> 3) + 8 * hi + (e & 7);
    Ap[t] = (_Float16)Wm[m * K + k];
}

// B (16-bit 32x16): lane l: N=l%16, hi=l/16; elem e: K = 16*(e/8)+8*hi+(e%8)
__global__ void pack_b_kernel(const float* __restrict__ T, _Float16* __restrict__ Bp,
                              int KC, int K, int total) {
    int t = blockIdx.x * blockDim.x + threadIdx.x;
    if (t >= total) return;
    int e = t & 15, l = (t >> 4) & 31, rest = t >> 9;
    int kc = rest % KC, tt = rest / KC;
    int r  = tt * 16 + (l & 15);
    int hi = l >> 4;
    int k  = kc * 32 + 16 * (e >> 3) + 8 * hi + (e & 7);
    Bp[t] = (_Float16)T[(size_t)r * K + k];
}

// ---------------- WMMA GEMM: Out[row][Cout], 8 waves = 128 channels / block ----------------
__global__ void gemm_wmma_kernel(const _Float16* __restrict__ Ap,
                                 const _Float16* __restrict__ Bp,
                                 float* __restrict__ Out, int KC, int Cout) {
    __shared__ float tile[16 * 128];
    int lane = threadIdx.x & 31;
    int wave = threadIdx.x >> 5;            // 0..7
    int mt   = blockIdx.y * 8 + wave;       // M-tile (16 channels)
    int tt   = blockIdx.x;                  // token tile (16 rows)
    const _Float16* ap = Ap + (size_t)(mt * KC) * 512 + lane * 16;
    const _Float16* bp = Bp + (size_t)(tt * KC) * 512 + lane * 16;
    v8f acc = {};
    for (int kc = 0; kc < KC; ++kc) {
        v16h a = *(const v16h*)(ap + kc * 512);
        v16h b = *(const v16h*)(bp + kc * 512);
        acc = __builtin_amdgcn_wmma_f32_16x16x32_f16(false, a, false, b,
                                                     (short)0, acc, false, false);
    }
    // C/D layout: elem r of v8f: M = r + 8*(lane/16), N = lane%16
    int tn = lane & 15, mh = (lane >> 4) * 8;
#pragma unroll
    for (int r = 0; r < 8; ++r) tile[tn * 128 + wave * 16 + mh + r] = acc[r];
    __syncthreads();
    int cb = blockIdx.y * 128, row0 = tt * 16;
    for (int i = threadIdx.x; i < 16 * 128; i += 256) {
        int rr = i >> 7, cc = i & 127;
        Out[(size_t)(row0 + rr) * Cout + cb + cc] = tile[i];
    }
}

// ---------------- BatchNorm (training stats) + exact GELU ----------------
__global__ void bn_stats_kernel(const float* __restrict__ X, int C,
                                float* __restrict__ gsum, float* __restrict__ gsq) {
    int ch = threadIdx.x;                   // blockDim == C
    int rpb = (RROWS + gridDim.x - 1) / gridDim.x;
    int r0 = blockIdx.x * rpb;
    int r1 = min(RROWS, r0 + rpb);
    float s = 0.f, q = 0.f;
    for (int r = r0; r < r1; ++r) {
        float v = X[(size_t)r * C + ch];
        s += v; q += v * v;
    }
    atomicAdd(&gsum[ch], s);
    atomicAdd(&gsq[ch], q);
}

// per-channel: scale = gamma*rsqrt(var+eps), shift = beta - mu*scale
__global__ void bn_finalize_kernel(const float* __restrict__ gsum, const float* __restrict__ gsq,
                                   const float* __restrict__ gam, const float* __restrict__ bet,
                                   float* __restrict__ scale, float* __restrict__ shift, int C) {
    int ch = blockIdx.x * blockDim.x + threadIdx.x;
    if (ch >= C) return;
    float mu  = gsum[ch] * INV_R;
    float var = gsq[ch] * INV_R - mu * mu;
    float sc  = gam[ch] * rsqrtf(var + BNEPS);
    scale[ch] = sc;
    shift[ch] = bet[ch] - mu * sc;
}

__global__ void bn_apply_gelu_kernel(const float* __restrict__ X, float* __restrict__ Y,
                                     int C, int total,
                                     const float* __restrict__ scale,
                                     const float* __restrict__ shift) {
    int i = blockIdx.x * blockDim.x + threadIdx.x;
    if (i >= total) return;
    int ch = i % C;
    float t = fmaf(X[i], scale[ch], shift[ch]);
    Y[i] = 0.5f * t * (1.0f + erff(t * 0.70710678118f));
}

// ---------------- point prep: pixel index, pixel counts, token denominators ----------------
__global__ void points_kernel(const float* __restrict__ loc, const int* __restrict__ idx_agg,
                              const float* __restrict__ aggw,
                              float* __restrict__ cnt, float* __restrict__ den,
                              int* __restrict__ pixidx) {
    int p = blockIdx.x * blockDim.x + threadIdx.x;
    if (p >= BATCH * NPTS) return;
    int b = p / NPTS;
    float lx = fminf(fmaxf(loc[2 * p + 0], -1.f), 1.f) * 0.5f + 0.5f;
    float ly = fminf(fmaxf(loc[2 * p + 1], -1.f), 1.f) * 0.5f + 0.5f;
    int gx = min(max(__float2int_rn(lx * (float)(W_ - 1)), 0), W_ - 1);
    int gy = min(max(__float2int_rn(ly * (float)(W_ - 1)), 0), W_ - 1);
    int pix = gy * W_ + gx;
    pixidx[p] = pix;
    atomicAdd(&cnt[b * HWPIX + pix], 1.0f);
    atomicAdd(&den[b * NTOK + idx_agg[p]], aggw[p]);
}

// y = 1/(x+eps)  (precompute reciprocals once instead of per-channel divides)
__global__ void invert_kernel(const float* __restrict__ X, float* __restrict__ Y, int n) {
    int i = blockIdx.x * blockDim.x + threadIdx.x;
    if (i >= n) return;
    Y[i] = 1.0f / (X[i] + EPSF);
}

// ---------------- token2map scatter (channel-major map for conv locality) ----------------
__global__ void scatter_kernel(const float* __restrict__ h, const int* __restrict__ idx_agg,
                               const int* __restrict__ pixidx, float* __restrict__ sums) {
    int t = blockIdx.x * blockDim.x + threadIdx.x;
    if (t >= BATCH * NPTS * CHID) return;
    int ch = t & 511, p = t >> 9;
    int b = p / NPTS;
    int row = b * NTOK + idx_agg[p];
    float v = h[(size_t)row * CHID + ch];
    atomicAdd(&sums[(size_t)(b * CHID + ch) * HWPIX + pixidx[p]], v);
}

// ---------------- fused: normalize map + depthwise 3x3 conv + map2token scatter ----------------
__global__ void convgather_kernel(const float* __restrict__ sums, const float* __restrict__ invcnt,
                                  const float* __restrict__ dww, const float* __restrict__ dwb,
                                  const int* __restrict__ idx_agg, const int* __restrict__ pixidx,
                                  const float* __restrict__ aggw, float* __restrict__ num) {
    int t = blockIdx.x * blockDim.x + threadIdx.x;
    if (t >= BATCH * NPTS * CHID) return;
    int ch = t & 511, p = t >> 9;
    int b = p / NPTS;
    int pix = pixidx[p];
    int gy = pix / W_, gx = pix % W_;
    const float* S  = sums + (size_t)(b * CHID + ch) * HWPIX;
    const float* IC = invcnt + b * HWPIX;
    float acc = dwb[ch];
#pragma unroll
    for (int ky = 0; ky < 3; ++ky) {
        int yy = gy + ky - 1;
        if (yy < 0 || yy >= W_) continue;
#pragma unroll
        for (int kx = 0; kx < 3; ++kx) {
            int xx = gx + kx - 1;
            if (xx < 0 || xx >= W_) continue;
            int q = yy * W_ + xx;
            acc = fmaf(dww[ch * 9 + ky * 3 + kx] * S[q], IC[q], acc);
        }
    }
    atomicAdd(&num[(size_t)(b * NTOK + idx_agg[p]) * CHID + ch], aggw[p] * acc);
}

// ---------------- skip: h = h*dws + num*invden ----------------
__global__ void combine_kernel(float* __restrict__ h, const float* __restrict__ num,
                               const float* __restrict__ invden, const float* __restrict__ dws) {
    int i = blockIdx.x * blockDim.x + threadIdx.x;
    if (i >= RROWS * CHID) return;
    int ch = i & 511, r = i >> 9;
    h[i] = fmaf(h[i], dws[ch], num[i] * invden[r]);
}

extern "C" void kernel_launch(void* const* d_in, const int* in_sizes, int n_in,
                              void* d_out, int out_size, void* d_ws, size_t ws_size,
                              hipStream_t stream) {
    const float* x        = (const float*)d_in[0];   // [B,N,128] token-major
    const float* loc_orig = (const float*)d_in[1];   // [B,N0,2]
    const int*   idx_agg  = (const int*)d_in[2];     // [B,N0]
    const float* agg_w    = (const float*)d_in[3];   // [B,N0]
    const float* fc1_w    = (const float*)d_in[4];   // [512,128]
    // fc1_b / fc2_b cancel under training-mode BN (shift removed by mean) -> skipped
    const float* g1       = (const float*)d_in[6];
    const float* b1       = (const float*)d_in[7];
    const float* dw_w     = (const float*)d_in[8];   // [512,1,3,3]
    const float* dw_b     = (const float*)d_in[9];
    const float* dws_w    = (const float*)d_in[10];
    const float* g2       = (const float*)d_in[11];
    const float* b2       = (const float*)d_in[12];
    const float* fc2_w    = (const float*)d_in[13];  // [128,512]
    const float* g3       = (const float*)d_in[15];
    const float* b3       = (const float*)d_in[16];
    float* out = (float*)d_out;                      // [B,N,128] token-major

    // ---- workspace carve-out ----
    char* ws = (char*)d_ws;
    size_t off = 0;
    auto carve = [&](size_t bytes) { char* p = ws + off; off = (off + bytes + 255) & ~(size_t)255; return p; };
    float*    h      = (float*)   carve((size_t)RROWS * CHID * 4);         // 25.7 MB
    float*    numb   = (float*)   carve((size_t)RROWS * CHID * 4);         // 25.7 MB
    float*    sums   = (float*)   carve((size_t)BATCH * CHID * HWPIX * 4); // 102.8 MB
    float*    cnt    = (float*)   carve((size_t)BATCH * HWPIX * 4);
    float*    invcnt = (float*)   carve((size_t)BATCH * HWPIX * 4);
    float*    den    = (float*)   carve((size_t)RROWS * 4);
    float*    invden = (float*)   carve((size_t)RROWS * 4);
    int*      pixidx = (int*)     carve((size_t)BATCH * NPTS * 4);
    float*    outpre = (float*)   carve((size_t)RROWS * COUT * 4);         // 6.4 MB
    _Float16* Ap1    = (_Float16*)carve((size_t)CHID * CIN * 2);
    _Float16* Bp1    = (_Float16*)carve((size_t)RROWS * CIN * 2);
    _Float16* Ap2    = (_Float16*)carve((size_t)COUT * CHID * 2);
    _Float16* Bp2    = (_Float16*)carve((size_t)RROWS * CHID * 2);
    float*    gsum   = (float*)   carve(CHID * 4);
    float*    gsq    = (float*)   carve(CHID * 4);
    float*    bscale = (float*)   carve(CHID * 4);
    float*    bshift = (float*)   carve(CHID * 4);

    // ---- zero the accumulation buffers (graph-capture-safe) ----
    hipMemsetAsync(sums, 0, (size_t)BATCH * CHID * HWPIX * 4, stream);
    hipMemsetAsync(cnt,  0, (size_t)BATCH * HWPIX * 4, stream);
    hipMemsetAsync(den,  0, (size_t)RROWS * 4, stream);
    hipMemsetAsync(numb, 0, (size_t)RROWS * CHID * 4, stream);

    const int TB = 256;
    // ---- fc1: pack + WMMA GEMM -> h [RROWS,512] ----
    {
        int tot_a = (CHID / 16) * (CIN / 32) * 512;   // 32*4*512
        pack_a_kernel<<<(tot_a + TB - 1) / TB, TB, 0, stream>>>(fc1_w, Ap1, CIN / 32, CIN, tot_a);
        int tot_b = (RROWS / 16) * (CIN / 32) * 512;  // 784*4*512
        pack_b_kernel<<<(tot_b + TB - 1) / TB, TB, 0, stream>>>(x, Bp1, CIN / 32, CIN, tot_b);
        gemm_wmma_kernel<<<dim3(RROWS / 16, CHID / 128), TB, 0, stream>>>(Ap1, Bp1, h, CIN / 32, CHID);
    }
    // ---- BN1 + GELU (in place) ----
    hipMemsetAsync(gsum, 0, CHID * 4, stream);
    hipMemsetAsync(gsq,  0, CHID * 4, stream);
    bn_stats_kernel<<<196, CHID, 0, stream>>>(h, CHID, gsum, gsq);
    bn_finalize_kernel<<<(CHID + TB - 1) / TB, TB, 0, stream>>>(gsum, gsq, g1, b1, bscale, bshift, CHID);
    bn_apply_gelu_kernel<<<(RROWS * CHID + TB - 1) / TB, TB, 0, stream>>>(
        h, h, CHID, RROWS * CHID, bscale, bshift);

    // ---- token2map scatter + fused dwconv + map2token ----
    points_kernel<<<(BATCH * NPTS + TB - 1) / TB, TB, 0, stream>>>(
        loc_orig, idx_agg, agg_w, cnt, den, pixidx);
    invert_kernel<<<(BATCH * HWPIX + TB - 1) / TB, TB, 0, stream>>>(cnt, invcnt, BATCH * HWPIX);
    invert_kernel<<<(RROWS + TB - 1) / TB, TB, 0, stream>>>(den, invden, RROWS);
    int tot_s = BATCH * NPTS * CHID;
    scatter_kernel<<<(tot_s + TB - 1) / TB, TB, 0, stream>>>(h, idx_agg, pixidx, sums);
    convgather_kernel<<<(tot_s + TB - 1) / TB, TB, 0, stream>>>(
        sums, invcnt, dw_w, dw_b, idx_agg, pixidx, agg_w, numb);

    // ---- skip combine, BN2 + GELU (in place) ----
    combine_kernel<<<(RROWS * CHID + TB - 1) / TB, TB, 0, stream>>>(h, numb, invden, dws_w);
    hipMemsetAsync(gsum, 0, CHID * 4, stream);
    hipMemsetAsync(gsq,  0, CHID * 4, stream);
    bn_stats_kernel<<<196, CHID, 0, stream>>>(h, CHID, gsum, gsq);
    bn_finalize_kernel<<<(CHID + TB - 1) / TB, TB, 0, stream>>>(gsum, gsq, g2, b2, bscale, bshift, CHID);
    bn_apply_gelu_kernel<<<(RROWS * CHID + TB - 1) / TB, TB, 0, stream>>>(
        h, h, CHID, RROWS * CHID, bscale, bshift);

    // ---- fc2: pack + WMMA GEMM -> outpre [RROWS,128] ----
    {
        int tot_a = (COUT / 16) * (CHID / 32) * 512;  // 8*16*512
        pack_a_kernel<<<(tot_a + TB - 1) / TB, TB, 0, stream>>>(fc2_w, Ap2, CHID / 32, CHID, tot_a);
        int tot_b = (RROWS / 16) * (CHID / 32) * 512; // 784*16*512
        pack_b_kernel<<<(tot_b + TB - 1) / TB, TB, 0, stream>>>(h, Bp2, CHID / 32, CHID, tot_b);
        gemm_wmma_kernel<<<dim3(RROWS / 16, 1), TB, 0, stream>>>(Ap2, Bp2, outpre, CHID / 32, COUT);
    }
    // ---- BN3 + GELU -> d_out ----
    hipMemsetAsync(gsum, 0, CHID * 4, stream);
    hipMemsetAsync(gsq,  0, CHID * 4, stream);
    bn_stats_kernel<<<196, COUT, 0, stream>>>(outpre, COUT, gsum, gsq);
    bn_finalize_kernel<<<(COUT + TB - 1) / TB, TB, 0, stream>>>(gsum, gsq, g3, b3, bscale, bshift, COUT);
    bn_apply_gelu_kernel<<<(RROWS * COUT + TB - 1) / TB, TB, 0, stream>>>(
        outpre, out, COUT, RROWS * COUT, bscale, bshift);
}
